// JetGNN_13228499272010
// MI455X (gfx1250) — matrix operand order repaired and checked by reading, not compiled, not run on hardware
//
#include <hip/hip_runtime.h>

typedef __attribute__((ext_vector_type(16))) _Float16 v16h;
typedef __attribute__((ext_vector_type(8)))  _Float16 v8h;
typedef __attribute__((ext_vector_type(4)))  _Float16 v4h;
typedef __attribute__((ext_vector_type(8)))  float    v8f;

#define NN 100000   // nodes
#define NE 3200000  // edges
#define NG 1024     // graphs

static inline int cdiv(long long a, long long b) { return (int)((a + b - 1) / b); }

// ---------------- small utility kernels ----------------

// d[n*FIN + k] = (f16) s[k*FOUT + n]   (transpose W for K-contiguous LDS staging)
__global__ void k_w_transpose_f16(const float* __restrict__ s, _Float16* __restrict__ d,
                                  int fin, int fout) {
  int idx = blockIdx.x * blockDim.x + threadIdx.x;
  if (idx < fin * fout) {
    int n = idx / fin, k = idx % fin;
    d[idx] = (_Float16)s[k * fout + n];
  }
}

__global__ void k_fill1(float* __restrict__ p, int n) {
  int i = blockIdx.x * blockDim.x + threadIdx.x;
  if (i < n) p[i] = 1.0f;   // self-loop contribution to degree
}

__global__ void k_deg(const int* __restrict__ col, float* __restrict__ deg, int E) {
  int e = blockIdx.x * blockDim.x + threadIdx.x;
  if (e < E) atomicAdd(&deg[col[e]], 1.0f);
}

__global__ void k_rsqrt(float* __restrict__ d, int n) {
  int i = blockIdx.x * blockDim.x + threadIdx.x;
  if (i < n) d[i] = rsqrtf(d[i]);   // deg >= 1 always (self loops)
}

// ---------------- WMMA GEMM: out[M,FOUT] = in[M,FIN] @ W[FIN,FOUT] ----------------
// wT is pre-transposed f16 weights [FOUT][FIN]. Persistent blocks: W staged in LDS
// once, then grid-stride over 16-row M tiles. One 16x16 tile per wave, FOUT/16 waves.
// Requires M % 16 == 0 (true here: 100000 and 1024).
// SCALE: multiply row m by scale[m].  BIASRELU: out = relu(out + bias[n]).
template <int FIN, int FOUT, int KP, bool SCALE, bool BIASRELU>
__global__ void __launch_bounds__((FOUT / 16) * 32)
k_gemm_wmma(const float* __restrict__ in, const _Float16* __restrict__ wT,
            const float* __restrict__ scale, const float* __restrict__ bias,
            float* __restrict__ out, int mtiles) {
  constexpr int WAVES = FOUT / 16;
  constexpr int NTHR  = WAVES * 32;
  __shared__ _Float16 ldsA[16 * KP];        // A tile, row-major [16][KP]
  __shared__ _Float16 ldsW[FOUT * KP];      // W transposed [FOUT][KP] (K contiguous)

  const int tid = threadIdx.y * 32 + threadIdx.x;

  // stage W once per block (4-half = 8B chunks)
  constexpr int CW = FIN / 4;
  for (int idx = tid; idx < FOUT * CW; idx += NTHR) {
    int n = idx / CW, c = idx % CW;
    *(v4h*)&ldsW[n * KP + c * 4] = *(const v4h*)&wT[n * FIN + c * 4];
  }
  // zero K-pad regions once (never overwritten afterwards)
  if (KP > FIN) {
    for (int idx = tid; idx < 16 * KP; idx += NTHR)
      if ((idx % KP) >= FIN) ldsA[idx] = (_Float16)0.0f;
    for (int idx = tid; idx < FOUT * KP; idx += NTHR)
      if ((idx % KP) >= FIN) ldsW[idx] = (_Float16)0.0f;
  }

  const int L     = threadIdx.x;  // lane 0..31 (wave32)
  const int wy    = threadIdx.y;  // N-tile index
  const int lhalf = L >> 4;       // 0: lanes 0-15, 1: lanes 16-31
  const int l15   = L & 15;

  for (int tile = blockIdx.x; tile < mtiles; tile += gridDim.x) {
    __syncthreads();  // W ready (1st iter) / previous iter's LDS reads done

    // stage A strip (16 x FIN), float4 global loads -> packed 4-half LDS stores
    constexpr int CA = FIN / 4;
    for (int idx = tid; idx < 16 * CA; idx += NTHR) {
      int r = idx / CA, c = idx % CA;
      const float4 v = *(const float4*)&in[(long long)(tile * 16 + r) * FIN + c * 4];
      v4h h;
      h[0] = (_Float16)v.x; h[1] = (_Float16)v.y;
      h[2] = (_Float16)v.z; h[3] = (_Float16)v.w;
      *(v4h*)&ldsA[r * KP + c * 4] = h;
    }
    __syncthreads();

    v8f acc = {};
    for (int kk = 0; kk < KP; kk += 32) {
      // A 16x32 f16 layout: lane half -> K base 0/8; elems 8..15 at +16
      const _Float16* pa = &ldsA[l15 * KP + kk + (lhalf << 3)];
      v8h a0 = *(const v8h*)pa;
      v8h a1 = *(const v8h*)(pa + 16);
      v16h a = __builtin_shufflevector(a0, a1, 0, 1, 2, 3, 4, 5, 6, 7,
                                       8, 9, 10, 11, 12, 13, 14, 15);
      // B 32x16 f16 layout: lane half -> K base 0/8; elems 8..15 at +16; N = lane&15
      const _Float16* pb = &ldsW[(wy * 16 + l15) * KP + kk + (lhalf << 3)];
      v8h b0 = *(const v8h*)pb;
      v8h b1 = *(const v8h*)(pb + 16);
      v16h b = __builtin_shufflevector(b0, b1, 0, 1, 2, 3, 4, 5, 6, 7,
                                       8, 9, 10, 11, 12, 13, 14, 15);
      acc = __builtin_amdgcn_wmma_f32_16x16x32_f16(false, a, false, b, (short)0, acc,
                                                   false, false);
    }

    // epilogue: C/D layout VGPR r -> M = tile*16 + r + 8*lhalf, N = wy*16 + (lane&15)
    const int rbase = tile * 16 + (lhalf << 3);
    float s[8];
    if (SCALE) {
      const float4 s0 = *(const float4*)&scale[rbase];
      const float4 s1 = *(const float4*)&scale[rbase + 4];
      s[0] = s0.x; s[1] = s0.y; s[2] = s0.z; s[3] = s0.w;
      s[4] = s1.x; s[5] = s1.y; s[6] = s1.z; s[7] = s1.w;
    }
    float bv = BIASRELU ? bias[wy * 16 + l15] : 0.0f;
    float* po = out + (long long)rbase * FOUT + wy * 16 + l15;
#pragma unroll
    for (int r = 0; r < 8; ++r) {
      float v = acc[r];
      if (SCALE) v *= s[r];
      if (BIASRELU) v = fmaxf(v + bv, 0.0f);
      po[r * FOUT] = v;
    }
  }
}

// ---------------- edge scatter: S[col] += Msg[row]  (float4 granularity) ----------------
template <int F>
__global__ void k_scatter(const float* __restrict__ Msg, const int* __restrict__ row,
                          const int* __restrict__ col, float* __restrict__ S, int E) {
  constexpr int C = F / 4;
  long long idx = (long long)blockIdx.x * blockDim.x + threadIdx.x;
  if (idx >= (long long)E * C) return;
  int e = (int)(idx / C);
  int c = (int)(idx % C);
  __builtin_prefetch(row + e + 4096, 0, 0);
  int r = row[e];
  int t = col[e];
  const float4 m = *reinterpret_cast<const float4*>(Msg + (long long)r * F + c * 4);
  float* dst = S + (long long)t * F + c * 4;
  atomicAdd(dst + 0, m.x);
  atomicAdd(dst + 1, m.y);
  atomicAdd(dst + 2, m.z);
  atomicAdd(dst + 3, m.w);
}

// ---------------- finalize: h = relu(dinv[i]*(S[i]+Msg[i]) + b) ----------------
template <int F>
__global__ void k_finalize(const float* __restrict__ S, const float* __restrict__ Msg,
                           const float* __restrict__ dinv, const float* __restrict__ bias,
                           float* __restrict__ out, int n) {
  constexpr int C = F / 4;
  long long idx = (long long)blockIdx.x * blockDim.x + threadIdx.x;
  if (idx >= (long long)n * C) return;
  int i = (int)(idx / C);
  int c = (int)(idx % C);
  float d = dinv[i];
  float4 s = *reinterpret_cast<const float4*>(S + (long long)i * F + c * 4);
  float4 m = *reinterpret_cast<const float4*>(Msg + (long long)i * F + c * 4);
  float4 o;
  o.x = fmaxf(d * (s.x + m.x) + bias[c * 4 + 0], 0.0f);
  o.y = fmaxf(d * (s.y + m.y) + bias[c * 4 + 1], 0.0f);
  o.z = fmaxf(d * (s.z + m.z) + bias[c * 4 + 2], 0.0f);
  o.w = fmaxf(d * (s.w + m.w) + bias[c * 4 + 3], 0.0f);
  *reinterpret_cast<float4*>(out + (long long)i * F + c * 4) = o;
}

// ---------------- global mean pool ----------------

__global__ void k_pool_cnt(const int* __restrict__ batch, float* __restrict__ cnts, int n) {
  int i = blockIdx.x * blockDim.x + threadIdx.x;
  if (i < n) atomicAdd(&cnts[batch[i]], 1.0f);
}

__global__ void k_pool_sum(const float* __restrict__ h, const int* __restrict__ batch,
                           float* __restrict__ sums, int n) {
  // F = 128, C = 32 float4 chunks
  long long idx = (long long)blockIdx.x * blockDim.x + threadIdx.x;
  if (idx >= (long long)n * 32) return;
  int i = (int)(idx >> 5);
  int c = (int)(idx & 31);
  int g = batch[i];
  const float4 v = *reinterpret_cast<const float4*>(h + (long long)i * 128 + c * 4);
  float* dst = sums + (long long)g * 128 + c * 4;
  atomicAdd(dst + 0, v.x);
  atomicAdd(dst + 1, v.y);
  atomicAdd(dst + 2, v.z);
  atomicAdd(dst + 3, v.w);
}

__global__ void k_pool_div(float* __restrict__ sums, const float* __restrict__ cnts, int n) {
  int idx = blockIdx.x * blockDim.x + threadIdx.x;  // n = NG*128
  if (idx < n) sums[idx] /= fmaxf(cnts[idx >> 7], 1.0f);
}

// ---------------- final tiny FC: out[g,j] = fc1[g,:] @ Wf2[:,j] + bf2[j] ----------------
__global__ void k_fc2(const float* __restrict__ fc1, const float* __restrict__ Wf2,
                      const float* __restrict__ bf2, float* __restrict__ out, int n) {
  int idx = blockIdx.x * blockDim.x + threadIdx.x;  // n = NG*2
  if (idx >= n) return;
  int g = idx >> 1, j = idx & 1;
  float acc = bf2[j];
#pragma unroll
  for (int k = 0; k < 64; ++k) acc += fc1[g * 64 + k] * Wf2[k * 2 + j];
  out[idx] = acc;
}

// ---------------- host side ----------------

extern "C" void kernel_launch(void* const* d_in, const int* in_sizes, int n_in,
                              void* d_out, int out_size, void* d_ws, size_t ws_size,
                              hipStream_t stream) {
  const float* x   = (const float*)d_in[0];
  const int*   ei  = (const int*)d_in[1];          // [2, NE]: row then col
  const int*   bat = (const int*)d_in[2];
  const float* W1  = (const float*)d_in[3];
  const float* b1  = (const float*)d_in[4];
  const float* W2  = (const float*)d_in[5];
  const float* b2  = (const float*)d_in[6];
  const float* W3  = (const float*)d_in[7];
  const float* b3  = (const float*)d_in[8];
  const float* Wf1 = (const float*)d_in[9];
  const float* bf1 = (const float*)d_in[10];
  const float* Wf2 = (const float*)d_in[11];
  const float* bf2 = (const float*)d_in[12];
  float* out = (float*)d_out;

  const int* row = ei;
  const int* col = ei + NE;

  // workspace carve-up (256B aligned)
  size_t off = 0;
  auto carve = [&](size_t bytes) {
    void* p = (char*)d_ws + off;
    off = (off + bytes + 255) & ~(size_t)255;
    return p;
  };
  float*    dinv = (float*)carve((size_t)NN * 4);
  float*    bufA = (float*)carve((size_t)NN * 128 * 4);  // layer io
  float*    bufM = (float*)carve((size_t)NN * 128 * 4);  // scaled messages
  float*    bufS = (float*)carve((size_t)NN * 128 * 4);  // scatter accumulator
  float*    sums = (float*)carve((size_t)NG * 128 * 4);
  float*    cnts = (float*)carve((size_t)NG * 4);
  float*    fc1  = (float*)carve((size_t)NG * 64 * 4);
  _Float16* w1t  = (_Float16*)carve((size_t)64 * 4 * 2);    // [64][4]
  _Float16* w2t  = (_Float16*)carve((size_t)64 * 64 * 2);   // [64][64]
  _Float16* w3t  = (_Float16*)carve((size_t)128 * 64 * 2);  // [128][64]
  _Float16* wf1t = (_Float16*)carve((size_t)64 * 128 * 2);  // [64][128]

  const int T = 256;
  const int MT = NN / 16;          // 6250 row tiles
  const int GEMM_GRID = 1024;      // persistent blocks

  // weights -> f16, transposed to [FOUT][FIN]
  k_w_transpose_f16<<<cdiv(4 * 64, T), T, 0, stream>>>(W1, w1t, 4, 64);
  k_w_transpose_f16<<<cdiv(64 * 64, T), T, 0, stream>>>(W2, w2t, 64, 64);
  k_w_transpose_f16<<<cdiv(64 * 128, T), T, 0, stream>>>(W3, w3t, 64, 128);
  k_w_transpose_f16<<<cdiv(128 * 64, T), T, 0, stream>>>(Wf1, wf1t, 128, 64);

  // symmetric normalization: dinv = rsqrt(1 + indegree)
  k_fill1<<<cdiv(NN, T), T, 0, stream>>>(dinv, NN);
  k_deg<<<cdiv(NE, T), T, 0, stream>>>(col, dinv, NE);
  k_rsqrt<<<cdiv(NN, T), T, 0, stream>>>(dinv, NN);

  // ---- layer 1: 4 -> 64 ----
  k_gemm_wmma<4, 64, 32, true, false>
      <<<GEMM_GRID, dim3(32, 4), 0, stream>>>(x, w1t, dinv, nullptr, bufM, MT);
  hipMemsetAsync(bufS, 0, (size_t)NN * 64 * 4, stream);
  k_scatter<64><<<cdiv((long long)NE * 16, T), T, 0, stream>>>(bufM, row, col, bufS, NE);
  k_finalize<64><<<cdiv((long long)NN * 16, T), T, 0, stream>>>(bufS, bufM, dinv, b1, bufA, NN);

  // ---- layer 2: 64 -> 64 ----
  k_gemm_wmma<64, 64, 64, true, false>
      <<<GEMM_GRID, dim3(32, 4), 0, stream>>>(bufA, w2t, dinv, nullptr, bufM, MT);
  hipMemsetAsync(bufS, 0, (size_t)NN * 64 * 4, stream);
  k_scatter<64><<<cdiv((long long)NE * 16, T), T, 0, stream>>>(bufM, row, col, bufS, NE);
  k_finalize<64><<<cdiv((long long)NN * 16, T), T, 0, stream>>>(bufS, bufM, dinv, b2, bufA, NN);

  // ---- layer 3: 64 -> 128 ----
  k_gemm_wmma<64, 128, 64, true, false>
      <<<GEMM_GRID, dim3(32, 8), 0, stream>>>(bufA, w3t, dinv, nullptr, bufM, MT);
  hipMemsetAsync(bufS, 0, (size_t)NN * 128 * 4, stream);
  k_scatter<128><<<cdiv((long long)NE * 32, T), T, 0, stream>>>(bufM, row, col, bufS, NE);
  k_finalize<128><<<cdiv((long long)NN * 32, T), T, 0, stream>>>(bufS, bufM, dinv, b3, bufA, NN);

  // ---- global mean pool ----
  hipMemsetAsync(sums, 0, (size_t)NG * 128 * 4, stream);
  hipMemsetAsync(cnts, 0, (size_t)NG * 4, stream);
  k_pool_cnt<<<cdiv(NN, T), T, 0, stream>>>(bat, cnts, NN);
  k_pool_sum<<<cdiv((long long)NN * 32, T), T, 0, stream>>>(bufA, bat, sums, NN);
  k_pool_div<<<cdiv(NG * 128, T), T, 0, stream>>>(sums, cnts, NG * 128);

  // ---- MLP head ----
  k_gemm_wmma<128, 64, 128, false, true>
      <<<NG / 16, dim3(32, 4), 0, stream>>>(sums, wf1t, nullptr, bf1, fc1, NG / 16);
  k_fc2<<<cdiv(NG * 2, T), T, 0, stream>>>(fc1, Wf2, bf2, out, NG * 2);
}
